// GRU_24438363914286
// MI455X (gfx1250) — matrix operand order repaired and checked by reading, not compiled, not run on hardware
//
#include <hip/hip_runtime.h>

typedef __attribute__((ext_vector_type(16))) __bf16 bf16x16;
typedef __attribute__((ext_vector_type(8)))  float  f32x8;
typedef __attribute__((ext_vector_type(4)))  unsigned int uint32x4;
typedef __attribute__((ext_vector_type(8)))  int    int32x8;
typedef __attribute__((ext_vector_type(4)))  int    int32x4;

#define TLEN  512
#define BATCH 64
#define EDIM  512
#define HDIM  1024
#define KTAG  64
#define G3H   3072
#define MROWS (TLEN * BATCH)   // 32768
#define KCHUNK 256             // K-chunk staged per TDM transfer in gru step

// ---------- bf16 helpers (storage = unsigned short, RNE convert) ----------
__device__ __forceinline__ unsigned short f2bf(float f) {
  unsigned int u = __float_as_uint(f);
  u += 0x7FFFu + ((u >> 16) & 1u);
  return (unsigned short)(u >> 16);
}
__device__ __forceinline__ float bf2f(unsigned short h) {
  return __uint_as_float(((unsigned int)h) << 16);
}

// A-fragment (16x32 bf16): lane holds row m0+(lane&15); K chunks
// [k+ksel .. k+ksel+7] and [k+16+ksel .. k+16+ksel+7], ksel = 8*(lane>=16).
__device__ __forceinline__ bf16x16 load_a_frag(const unsigned short* p) {
  union { bf16x16 v; unsigned short u[16]; } r;
#pragma unroll
  for (int i = 0; i < 8; ++i) { r.u[i] = p[i]; r.u[i + 8] = p[i + 16]; }
  return r.v;
}
// B-fragment (32x16 bf16): lane holds col n0+(lane&15); K chunks
// [k+ksel .. k+ksel+7] and [k+16+ksel .. k+16+ksel+7], ksel = 8*(lane>=16).
__device__ __forceinline__ bf16x16 load_b_frag(const unsigned short* p) {
  union { bf16x16 v; unsigned short u[16]; } r;
#pragma unroll
  for (int i = 0; i < 8; ++i) { r.u[i] = p[i]; r.u[i + 8] = p[i + 16]; }
  return r.v;
}

// ---------- Tensor Data Mover: 2D tile (rows x tile_cols bf16) -> LDS ----------
// D# per CDNA5 ISA ch.8: group0 = {flags, lds_addr, global_addr, type=2},
// group1 = {data_size=2B, tensor_dim0/1, tile_dim0/1, dim0_stride}.
// This toolchain exposes the 6-arg builtin: (g0, g1, g2, g3, g4, cpol).
__device__ __forceinline__ void tdm_load_2d_bf16(const unsigned short* gaddr,
                                                 unsigned lds_off,
                                                 unsigned tile_rows,
                                                 unsigned tile_cols,
                                                 unsigned tensor_cols) {
  unsigned long long ga = (unsigned long long)(const void*)gaddr;
  uint32x4 g0;
  g0[0] = 1u;                                        // count=1, user descriptor
  g0[1] = lds_off;                                   // LDS byte address
  g0[2] = (unsigned)(ga & 0xFFFFFFFFu);              // global_addr[31:0]
  g0[3] = (unsigned)((ga >> 32) & 0x01FFFFFFu) | (2u << 30); // addr[56:32] | type=2
  int32x8 g1;
  g1[0] = (int)(1u << 16);                           // data_size = 1 (2 bytes)
  g1[1] = (int)((tensor_cols & 0xFFFFu) << 16);      // tensor_dim0[15:0] @ bits 63:48
  g1[2] = (int)(((tensor_cols >> 16) & 0xFFFFu) |    // tensor_dim0[31:16]
                ((tile_rows & 0xFFFFu) << 16));      // tensor_dim1[15:0]
  g1[3] = (int)(((tile_rows >> 16) & 0xFFFFu) |      // tensor_dim1[31:16]
                ((tile_cols & 0xFFFFu) << 16));      // tile_dim0
  g1[4] = (int)(tile_rows & 0xFFFFu);                // tile_dim1 (tile_dim2 = 0)
  g1[5] = (int)tensor_cols;                          // tensor_dim0_stride[31:0]
  g1[6] = 0;                                         // stride0[47:32], stride1 lo
  g1[7] = 0;
  int32x4 z4 = {0, 0, 0, 0};
  int32x8 z8 = {0, 0, 0, 0, 0, 0, 0, 0};
  __builtin_amdgcn_tensor_load_to_lds(g0, g1, z4, z4, z8, 0);
}

// ---------------- setup kernels ----------------
__global__ void k_convert(const float* __restrict__ s, unsigned short* __restrict__ d, int n) {
  int i = blockIdx.x * blockDim.x + threadIdx.x;
  if (i < n) d[i] = f2bf(s[i]);
}

__global__ void k_gather(const int* __restrict__ x, const float* __restrict__ emb,
                         unsigned short* __restrict__ xe) {
  int idx = blockIdx.x * blockDim.x + threadIdx.x;       // < T*B*E = 16777216
  int row = idx >> 9;            // t*64 + b
  int e   = idx & 511;
  int t   = row >> 6;
  int b   = row & 63;
  int tok = x[b * TLEN + t];
  xe[idx] = f2bf(emb[(size_t)tok * EDIM + e]);
}

__global__ void k_zero(float* __restrict__ h32, unsigned short* __restrict__ hpp, int n) {
  int i = blockIdx.x * blockDim.x + threadIdx.x;
  if (i < n) { h32[i] = 0.0f; hpp[i] = 0; }
}

// ---------------- gi = xe @ w_ih^T + b_ih  (M=32768, N=3072, K=512) ----------------
__global__ void k_gemm_gi(const unsigned short* __restrict__ xe,
                          const unsigned short* __restrict__ wih,
                          const float* __restrict__ bih,
                          unsigned short* __restrict__ gi) {
  int lane = threadIdx.x & 31;
  int wave = threadIdx.x >> 5;
  int tile = blockIdx.x * 8 + wave;          // 393216 tiles exact
  int nt = tile % 192, mt = tile / 192;
  int m0 = mt * 16, n0 = nt * 16;
  const unsigned short* A  = xe  + (size_t)(m0 + (lane & 15)) * EDIM + ((lane >> 4) << 3);
  const unsigned short* Bp = wih + (size_t)(n0 + (lane & 15)) * EDIM + ((lane >> 4) << 3);
  f32x8 c = {};
#pragma unroll 4
  for (int k = 0; k < EDIM; k += 32) {
    bf16x16 a = load_a_frag(A + k);
    bf16x16 b = load_b_frag(Bp + k);
    c = __builtin_amdgcn_wmma_f32_16x16x32_bf16(false, a, false, b, (short)0, c, false, false);
  }
  int n = n0 + (lane & 15);
  float bias = bih[n];
  int mb = m0 + ((lane >> 4) << 3);
#pragma unroll
  for (int r = 0; r < 8; ++r)
    gi[(size_t)(mb + r) * G3H + n] = f2bf(c[r] + bias);
}

// ---------------- one GRU timestep ----------------
// 16 blocks x 256 threads; block owns j-slice [j0, j0+64).
// TDM double-buffers h_prev (64 x KCHUNK bf16 per chunk) into LDS while the
// 8 waves run 3 gates x 4 Mtiles x 4 Ntiles = 48 WMMA tiles out of LDS.
__global__ void k_gru_step(const unsigned short* __restrict__ gi,
                           const unsigned short* __restrict__ whh,
                           const float* __restrict__ bhh,
                           const unsigned short* __restrict__ hprev_bf,
                           const float* __restrict__ h32prev,
                           float* __restrict__ h32cur,
                           unsigned short* __restrict__ hcur_bf,
                           unsigned short* __restrict__ hs_bf,
                           int t) {
  __shared__ unsigned short shA[2][BATCH * KCHUNK];  // 2 x 32 KB
  __shared__ float ghs[3 * 64 * 64];                 // 48 KB
  int lane = threadIdx.x & 31;
  int wave = threadIdx.x >> 5;
  int j0 = blockIdx.x * 64;
  const int NCH = HDIM / KCHUNK;                     // 4 chunks

  unsigned shA_lds = (unsigned)(size_t)(const void*)&shA[0][0];  // flat addr[31:0] = LDS offset

  f32x8 acc[6];
#pragma unroll
  for (int i = 0; i < 6; ++i) acc[i] = (f32x8){};

  if (wave == 0 && lane == 0)                        // prefetch chunk 0
    tdm_load_2d_bf16(hprev_bf, shA_lds, BATCH, KCHUNK, HDIM);

  for (int ko = 0; ko < NCH; ++ko) {
    if (wave == 0 && lane == 0) {
      if (ko + 1 < NCH)                              // prefetch next chunk
        tdm_load_2d_bf16(hprev_bf + (ko + 1) * KCHUNK,
                         shA_lds + ((ko + 1) & 1) * (BATCH * KCHUNK * 2),
                         BATCH, KCHUNK, HDIM);
      // tensor ops complete in-order per wave: <=1 outstanding => chunk ko done
      if (ko + 1 < NCH) __builtin_amdgcn_s_wait_tensorcnt(1);
      else              __builtin_amdgcn_s_wait_tensorcnt(0);
    }
    __syncthreads();                                 // chunk ko visible to all waves
    const unsigned short* shAbuf = &shA[ko & 1][0];
#pragma unroll
    for (int i = 0; i < 6; ++i) {
      int tile = i * 8 + wave;                       // 0..47
      int gate = tile >> 4;
      int rem  = tile & 15;
      int mt = rem >> 2, nt = rem & 3;
      const unsigned short* Ap = shAbuf + (mt * 16 + (lane & 15)) * KCHUNK + ((lane >> 4) << 3);
      const unsigned short* Bp = whh +
          (size_t)(gate * HDIM + j0 + nt * 16 + (lane & 15)) * HDIM +
          ko * KCHUNK + ((lane >> 4) << 3);
#pragma unroll 2
      for (int k = 0; k < KCHUNK; k += 32) {
        bf16x16 a = load_a_frag(Ap + k);
        bf16x16 b = load_b_frag(Bp + k);
        acc[i] = __builtin_amdgcn_wmma_f32_16x16x32_bf16(false, a, false, b, (short)0,
                                                         acc[i], false, false);
      }
    }
    __syncthreads();                                 // done reading before next overwrite
  }

#pragma unroll
  for (int i = 0; i < 6; ++i) {
    int tile = i * 8 + wave;
    int gate = tile >> 4;
    int rem  = tile & 15;
    int mt = rem >> 2, nt = rem & 3;
    int nloc = nt * 16 + (lane & 15);
    int mb = mt * 16 + ((lane >> 4) << 3);
#pragma unroll
    for (int r = 0; r < 8; ++r)
      ghs[gate * 4096 + (mb + r) * 64 + nloc] = acc[i][r];
  }
  __syncthreads();

#pragma unroll 4
  for (int i = 0; i < 16; ++i) {
    int idx = threadIdx.x + 256 * i;     // 0..4095
    int b  = idx >> 6;
    int jj = idx & 63;
    int j  = j0 + jj;
    size_t girow = ((size_t)t * BATCH + b) * G3H;
    float i_r = bf2f(gi[girow + j]);
    float i_z = bf2f(gi[girow + HDIM + j]);
    float i_n = bf2f(gi[girow + 2 * HDIM + j]);
    float h_r = ghs[idx]        + bhh[j];
    float h_z = ghs[4096 + idx] + bhh[HDIM + j];
    float h_n = ghs[8192 + idx] + bhh[2 * HDIM + j];
    float r = 1.0f / (1.0f + __expf(-(i_r + h_r)));
    float z = 1.0f / (1.0f + __expf(-(i_z + h_z)));
    float nn = tanhf(i_n + r * h_n);
    float hp = h32prev[b * HDIM + j];
    float hnew = (1.0f - z) * nn + z * hp;
    h32cur[b * HDIM + j] = hnew;
    unsigned short hb = f2bf(hnew);
    hcur_bf[b * HDIM + j] = hb;
    hs_bf[((size_t)t * BATCH + b) * HDIM + j] = hb;
  }
}

// ---------------- emissions = hs @ fc_w^T + fc_b (M=32768, N=64, K=1024) ----------------
// output layout em[b][t][k]  (b*T*K + t*K + k) for the per-batch CRF scans.
__global__ void k_gemm_em(const unsigned short* __restrict__ hs_bf,
                          const unsigned short* __restrict__ fcw,
                          const float* __restrict__ fcb,
                          float* __restrict__ em) {
  int lane = threadIdx.x & 31;
  int wave = threadIdx.x >> 5;
  int tile = blockIdx.x * 8 + wave;       // 8192 tiles exact
  int nt = tile & 3, mt = tile >> 2;
  int m0 = mt * 16, n0 = nt * 16;
  const unsigned short* A  = hs_bf + (size_t)(m0 + (lane & 15)) * HDIM + ((lane >> 4) << 3);
  const unsigned short* Bp = fcw   + (size_t)(n0 + (lane & 15)) * HDIM + ((lane >> 4) << 3);
  f32x8 c = {};
  for (int k = 0; k < HDIM; k += 32) {
    bf16x16 a = load_a_frag(A + k);
    bf16x16 b = load_b_frag(Bp + k);
    c = __builtin_amdgcn_wmma_f32_16x16x32_bf16(false, a, false, b, (short)0, c, false, false);
  }
  int n = n0 + (lane & 15);
  float bias = fcb[n];
  int mb = m0 + ((lane >> 4) << 3);
#pragma unroll
  for (int r = 0; r < 8; ++r) {
    int mrow = mb + r;                 // = t*64 + b
    int tt = mrow >> 6;
    int bb = mrow & 63;
    em[((size_t)bb << 15) + (tt << 6) + n] = c[r] + bias;
  }
}

// ---------------- CRF numerator (gold path score), thread per batch ----------------
__global__ void k_crf_num(const int* __restrict__ tags, const float* __restrict__ em,
                          const float* __restrict__ start, const float* __restrict__ endv,
                          const float* __restrict__ trans, float* __restrict__ numo) {
  int b = threadIdx.x;
  if (b >= BATCH) return;
  int tg = tags[b * TLEN];
  float acc = start[tg] + em[((size_t)b << 15) + tg];
  for (int t = 1; t < TLEN; ++t) {
    int tc = tags[b * TLEN + t];
    acc += trans[tg * KTAG + tc] + em[((size_t)b << 15) + (t << 6) + tc];
    tg = tc;
  }
  acc += endv[tg];
  numo[b] = acc;
}

// ---------------- CRF forward (log partition), block per batch ----------------
__global__ void k_crf_fwd(const float* __restrict__ em, const float* __restrict__ start,
                          const float* __restrict__ endv, const float* __restrict__ trans,
                          float* __restrict__ deno) {
  __shared__ float tr[KTAG * KTAG];
  __shared__ float alpha[KTAG];
  __shared__ float red[KTAG];
  int b = blockIdx.x;
  int j = threadIdx.x;
  for (int i = j; i < KTAG * KTAG; i += KTAG) tr[i] = trans[i];
  alpha[j] = start[j] + em[((size_t)b << 15) + j];
  __syncthreads();
  for (int t = 1; t < TLEN; ++t) {
    float mx = -3.0e38f;
#pragma unroll 8
    for (int i = 0; i < KTAG; ++i) mx = fmaxf(mx, alpha[i] + tr[i * KTAG + j]);
    float s = 0.0f;
#pragma unroll 8
    for (int i = 0; i < KTAG; ++i) s += __expf(alpha[i] + tr[i * KTAG + j] - mx);
    float an = mx + __logf(s) + em[((size_t)b << 15) + (t << 6) + j];
    __syncthreads();
    alpha[j] = an;
    __syncthreads();
  }
  red[j] = alpha[j] + endv[j];
  __syncthreads();
  if (j == 0) {
    float mx = red[0];
    for (int i = 1; i < KTAG; ++i) mx = fmaxf(mx, red[i]);
    float s = 0.0f;
    for (int i = 0; i < KTAG; ++i) s += __expf(red[i] - mx);
    deno[b] = mx + __logf(s);
  }
}

__global__ void k_final(const float* __restrict__ deno, const float* __restrict__ numo,
                        float* __restrict__ out) {
  if (threadIdx.x == 0) {
    float s = 0.0f;
    for (int b = 0; b < BATCH; ++b) s += deno[b] - numo[b];
    out[0] = s;
  }
}

// ---------------- host orchestration ----------------
extern "C" void kernel_launch(void* const* d_in, const int* in_sizes, int n_in,
                              void* d_out, int out_size, void* d_ws, size_t ws_size,
                              hipStream_t stream) {
  (void)in_sizes; (void)n_in; (void)out_size; (void)ws_size;
  const int*   x    = (const int*)d_in[0];
  const int*   tags = (const int*)d_in[1];
  const float* emb  = (const float*)d_in[2];
  const float* w_ih = (const float*)d_in[3];
  const float* w_hh = (const float*)d_in[4];
  const float* b_ih = (const float*)d_in[5];
  const float* b_hh = (const float*)d_in[6];
  const float* fc_w = (const float*)d_in[7];
  const float* fc_b = (const float*)d_in[8];
  const float* stt  = (const float*)d_in[9];
  const float* ent  = (const float*)d_in[10];
  const float* trn  = (const float*)d_in[11];

  char* ws = (char*)d_ws;
  size_t off = 0;
  auto alloc = [&](size_t bytes) -> char* {
    char* p = ws + off;
    off = (off + bytes + 255) & ~(size_t)255;
    return p;
  };
  unsigned short* xe_bf  = (unsigned short*)alloc((size_t)MROWS * EDIM * 2);
  unsigned short* gi_bf  = (unsigned short*)alloc((size_t)MROWS * G3H * 2);
  unsigned short* hs_bf  = (unsigned short*)alloc((size_t)MROWS * HDIM * 2);
  unsigned short* hpp_bf = (unsigned short*)alloc((size_t)2 * BATCH * HDIM * 2);
  float*          h32    = (float*)alloc((size_t)2 * BATCH * HDIM * 4);
  unsigned short* wih_bf = (unsigned short*)alloc((size_t)G3H * EDIM * 2);
  unsigned short* whh_bf = (unsigned short*)alloc((size_t)G3H * HDIM * 2);
  unsigned short* fcw_bf = (unsigned short*)alloc((size_t)KTAG * HDIM * 2);
  float*          em     = (float*)alloc((size_t)BATCH * TLEN * KTAG * 4);
  float*          numo   = (float*)alloc(BATCH * 4);
  float*          deno   = (float*)alloc(BATCH * 4);

  const int BH = BATCH * HDIM;

  // 1. convert weights to bf16
  k_convert<<<(G3H * EDIM + 255) / 256, 256, 0, stream>>>(w_ih, wih_bf, G3H * EDIM);
  k_convert<<<(G3H * HDIM + 255) / 256, 256, 0, stream>>>(w_hh, whh_bf, G3H * HDIM);
  k_convert<<<(KTAG * HDIM + 255) / 256, 256, 0, stream>>>(fc_w, fcw_bf, KTAG * HDIM);
  // 2. embedding gather -> xe[t][b][e] bf16
  k_gather<<<(MROWS * EDIM) / 256, 256, 0, stream>>>(x, emb, xe_bf);
  // 3. zero initial hidden (both ping-pong buffers)
  k_zero<<<(2 * BH + 255) / 256, 256, 0, stream>>>(h32, hpp_bf, 2 * BH);
  // 4. big input-projection GEMM (bf16 WMMA)
  k_gemm_gi<<<(2048 * 192) / 8, 256, 0, stream>>>(xe_bf, wih_bf, b_ih, gi_bf);
  // 5. sequential GRU scan, one kernel per timestep (TDM-staged A operand)
  for (int t = 0; t < TLEN; ++t) {
    int cur = t & 1, prev = cur ^ 1;
    k_gru_step<<<HDIM / 64, 256, 0, stream>>>(
        gi_bf, whh_bf, b_hh,
        hpp_bf + (size_t)prev * BH, h32 + (size_t)prev * BH,
        h32 + (size_t)cur * BH, hpp_bf + (size_t)cur * BH,
        hs_bf, t);
  }
  // 6. emissions GEMM (bf16 WMMA)
  k_gemm_em<<<(2048 * 4) / 8, 256, 0, stream>>>(hs_bf, fcw_bf, fc_b, em);
  // 7. CRF
  k_crf_num<<<1, 64, 0, stream>>>(tags, em, stt, ent, trn, numo);
  k_crf_fwd<<<BATCH, KTAG, 0, stream>>>(em, stt, ent, trn, deno);
  k_final<<<1, 32, 0, stream>>>(deno, numo, (float*)d_out);
}
